// RNNCellModel_24756191494435
// MI455X (gfx1250) — matrix operand order repaired and checked by reading, not compiled, not run on hardware
//
#include <hip/hip_runtime.h>
#include <stdint.h>
#include <stddef.h>

typedef __bf16 bf16;
typedef __attribute__((ext_vector_type(8)))  bf16  v8bf;
typedef __attribute__((ext_vector_type(16))) bf16  v16bf;
typedef __attribute__((ext_vector_type(8)))  float v8f;

// ---------------- utility kernels ----------------

__global__ void k_cvt_f32_bf16(bf16* __restrict__ dst, const float* __restrict__ src, int n) {
  int i = blockIdx.x * blockDim.x + threadIdx.x;
  if (i < n) dst[i] = (bf16)src[i];
}

__global__ void k_zero_bf16(bf16* __restrict__ dst, int n) {
  int i = blockIdx.x * blockDim.x + threadIdx.x;
  if (i < n) dst[i] = (bf16)0.0f;
}

__global__ void k_bias_sum(float* __restrict__ dst, const float* __restrict__ a,
                           const float* __restrict__ b, int n) {
  int i = blockIdx.x * blockDim.x + threadIdx.x;
  if (i < n) dst[i] = a[i] + b[i];
}

// Wcat[h][k] = k<IN ? Wih[h][k] : (k<IN+H ? Whh[h][k-IN] : 0)   (fp32 -> bf16)
__global__ void k_pack_weights(bf16* __restrict__ dst, const float* __restrict__ wih,
                               const float* __restrict__ whh, int H, int IN, int Kpad) {
  long long id = (long long)blockIdx.x * blockDim.x + threadIdx.x;
  if (id >= (long long)H * Kpad) return;
  int h = (int)(id / Kpad);
  int k = (int)(id % Kpad);
  float v = 0.0f;
  if (k < IN)          v = wih[(long long)h * IN + k];
  else if (k < IN + H) v = whh[(long long)h * H + (k - IN)];
  dst[id] = (bf16)v;
}

// Concatenate up to 5 bf16 segments (per-batch-row strides) into act[16][Kpad].
// Columns [Ktot, Kpad) were zeroed once at call start and are never written.
__global__ void k_pack_act(bf16* __restrict__ dst, int Kpad, int Ktot,
                           const bf16* __restrict__ s0, int t0, int w0,
                           const bf16* __restrict__ s1, int t1, int w1,
                           const bf16* __restrict__ s2, int t2, int w2,
                           const bf16* __restrict__ s3, int t3, int w3,
                           const bf16* __restrict__ s4, int t4, int w4) {
  int id = blockIdx.x * blockDim.x + threadIdx.x;
  if (id >= 16 * Ktot) return;
  int b   = id / Ktot;
  int col = id % Ktot;
  int c   = col;
  bf16 v;
  if (c < w0) v = s0[b * t0 + c];
  else {
    c -= w0;
    if (c < w1) v = s1[b * t1 + c];
    else {
      c -= w1;
      if (c < w2) v = s2[b * t2 + c];
      else {
        c -= w2;
        if (c < w3) v = s3[b * t3 + c];
        else {
          c -= w3;
          if (c < w4) v = s4[b * t4 + c];
          else        v = (bf16)0.0f;
        }
      }
    }
  }
  dst[b * Kpad + col] = v;
}

// ---------------- WMMA split-K GEMM, dual independent K streams ----------------
// part[(s*16 + m)*H + n] = sum over this split's K-range of act[m][k] * Wcat[n][k]
// One wave handles 3 adjacent 16-wide N tiles (H % 48 == 0 for 4800 and 1200).
// The wave's K range is processed as TWO independent streams with separate
// accumulators/registers so loads of one stream overlap WMMAs of the other.

#define NTLOAD(p) __builtin_nontemporal_load((const v8bf*)(p))
#define CAT16(lo, hi) __builtin_shufflevector(lo, hi, 0,1,2,3,4,5,6,7,8,9,10,11,12,13,14,15)
#define WMMA_BF16(A, B, C) \
  __builtin_amdgcn_wmma_f32_16x16x32_bf16(false, A, false, B, (short)0, C, false, false)

__global__ void __launch_bounds__(128)
k_cell_gemm_splitk(const bf16* __restrict__ act, const bf16* __restrict__ wcat,
                   int Kpad, int H, int S, float* __restrict__ part) {
  const int wavesPerBlk = blockDim.x >> 5;
  const int wave = blockIdx.x * wavesPerBlk + (threadIdx.x >> 5);
  const int tileGroups = H / 48;
  const int s = wave / tileGroups;
  if (s >= S) return;
  const int g = wave % tileGroups;
  const int tileBase = g * 48;

  // contiguous K range for this split, multiple of 32, always non-empty
  const int chunks = Kpad >> 5;
  const int cps = (chunks + S - 1) / S;
  int kc0 = s * cps * 32;
  int kc1 = kc0 + cps * 32;
  if (kc1 > Kpad) kc1 = Kpad;

  const int lane = threadIdx.x & 31;
  const int row  = lane & 15;
  const int hi   = lane >> 4;
  // A frag (16x32 bf16): lanes 0-15 hold K 0..15, lanes 16-31 hold K 16..31
  const int aLo  = hi ? 16 : 0;
  const int aHi  = hi ? 24 : 8;
  // B frag (32x16 bf16): lane = column; lanes 0-15 hold K 0..15, lanes 16-31 hold K 16..31
  const int bOff = hi ? 16 : 0;

  const bf16* arow = act  + (size_t)row * Kpad;
  const bf16* w0r  = wcat + (size_t)(tileBase +      row) * Kpad;
  const bf16* w1r  = wcat + (size_t)(tileBase + 16 + row) * Kpad;
  const bf16* w2r  = wcat + (size_t)(tileBase + 32 + row) * Kpad;

  // split this wave's chunks into two independent streams
  const int nch = (kc1 - kc0) >> 5;
  const int nB  = nch >> 1;        // stream-B chunk count
  const int nA  = nch - nB;        // stream-A chunk count (nA == nB or nB+1)
  const int kB0 = kc0 + nA * 32;

  v8f acc0a = {}, acc1a = {}, acc2a = {};
  v8f acc0b = {}, acc1b = {}, acc2b = {};

  for (int i = 0; i < nB; ++i) {
    const int kA = kc0 + i * 32;
    const int kB = kB0 + i * 32;

    // stream A loads
    v8bf aa0  = *(const v8bf*)(arow + kA + aLo);
    v8bf aa1  = *(const v8bf*)(arow + kA + aHi);
    v8bf ab00 = NTLOAD(w0r + kA + bOff);
    v8bf ab01 = NTLOAD(w0r + kA + bOff + 8);
    v8bf ab10 = NTLOAD(w1r + kA + bOff);
    v8bf ab11 = NTLOAD(w1r + kA + bOff + 8);
    v8bf ab20 = NTLOAD(w2r + kA + bOff);
    v8bf ab21 = NTLOAD(w2r + kA + bOff + 8);

    // stream B loads (independent registers, independent addresses)
    v8bf ba0  = *(const v8bf*)(arow + kB + aLo);
    v8bf ba1  = *(const v8bf*)(arow + kB + aHi);
    v8bf bb00 = NTLOAD(w0r + kB + bOff);
    v8bf bb01 = NTLOAD(w0r + kB + bOff + 8);
    v8bf bb10 = NTLOAD(w1r + kB + bOff);
    v8bf bb11 = NTLOAD(w1r + kB + bOff + 8);
    v8bf bb20 = NTLOAD(w2r + kB + bOff);
    v8bf bb21 = NTLOAD(w2r + kB + bOff + 8);

    // stream A compute (stream B loads still in flight)
    v16bf Aa = CAT16(aa0, aa1);
    acc0a = WMMA_BF16(Aa, CAT16(ab00, ab01), acc0a);
    acc1a = WMMA_BF16(Aa, CAT16(ab10, ab11), acc1a);
    acc2a = WMMA_BF16(Aa, CAT16(ab20, ab21), acc2a);

    // stream B compute
    v16bf Ab = CAT16(ba0, ba1);
    acc0b = WMMA_BF16(Ab, CAT16(bb00, bb01), acc0b);
    acc1b = WMMA_BF16(Ab, CAT16(bb10, bb11), acc1b);
    acc2b = WMMA_BF16(Ab, CAT16(bb20, bb21), acc2b);
  }

  if (nA > nB) {  // at most one leftover chunk, handled by stream A
    const int kA = kc0 + nB * 32;
    v8bf aa0  = *(const v8bf*)(arow + kA + aLo);
    v8bf aa1  = *(const v8bf*)(arow + kA + aHi);
    v8bf ab00 = NTLOAD(w0r + kA + bOff);
    v8bf ab01 = NTLOAD(w0r + kA + bOff + 8);
    v8bf ab10 = NTLOAD(w1r + kA + bOff);
    v8bf ab11 = NTLOAD(w1r + kA + bOff + 8);
    v8bf ab20 = NTLOAD(w2r + kA + bOff);
    v8bf ab21 = NTLOAD(w2r + kA + bOff + 8);
    v16bf Aa = CAT16(aa0, aa1);
    acc0a = WMMA_BF16(Aa, CAT16(ab00, ab01), acc0a);
    acc1a = WMMA_BF16(Aa, CAT16(ab10, ab11), acc1a);
    acc2a = WMMA_BF16(Aa, CAT16(ab20, ab21), acc2a);
  }

  v8f acc0 = acc0a + acc0b;
  v8f acc1 = acc1a + acc1b;
  v8f acc2 = acc2a + acc2b;

  // C/D layout: d[r] at lane l -> out[M = r + 8*(l>=16)][N = l%16]
  float* p = part + (size_t)(s * 16) * H;
#pragma unroll
  for (int r = 0; r < 8; ++r) {
    int m = r + (hi << 3);
    p[(size_t)m * H + tileBase +      row] = acc0[r];
    p[(size_t)m * H + tileBase + 16 + row] = acc1[r];
    p[(size_t)m * H + tileBase + 32 + row] = acc2[r];
  }
}

// ---------------- epilogue: reduce splits, bias, tanh, write traj + bf16 hstate ----------------

static __device__ __forceinline__ float fast_tanh(float x) {
  float a = __builtin_fabsf(x);
  float e = __expf(2.0f * a);
  float r = 1.0f - 2.0f / (e + 1.0f);   // == tanh(|x|), robust for large |x| (e -> inf)
  return __builtin_copysignf(r, x);
}

__global__ void k_cell_epilogue(const float* __restrict__ part, const float* __restrict__ bias,
                                int H, int S, float* __restrict__ out, int outBatchStride,
                                bf16* __restrict__ hstate) {
  int id = blockIdx.x * blockDim.x + threadIdx.x;
  if (id >= 16 * H) return;
  int m = id / H;
  int n = id % H;
  float acc = bias[n];
  for (int s = 0; s < S; ++s) acc += part[(size_t)(s * 16 + m) * H + n];
  float h = fast_tanh(acc);
  out[(size_t)m * outBatchStride + n] = h;
  hstate[id] = (bf16)h;
}

// ---------------- host ----------------

static inline int cdiv(int a, int b) { return (a + b - 1) / b; }

extern "C" void kernel_launch(void* const* d_in, const int* in_sizes, int n_in,
                              void* d_out, int out_size, void* d_ws, size_t ws_size,
                              hipStream_t stream) {
  (void)in_sizes; (void)n_in; (void)out_size; (void)ws_size;
  constexpr int Bn = 16, T = 50, X = 3600, E4 = 4800, I4 = 1200, E23 = 4800, I23 = 1200;
  constexpr int IN4E = 2*X + I4 + E23;   // 13200
  constexpr int IN4I = 2*X + E4 + E23;   // 16800
  constexpr int IN23E = E4 + I23;        // 6000
  constexpr int IN23I = E4 + E23;        // 9600
  constexpr int K4E = IN4E + E4, K4I = IN4I + I4, K23E = IN23E + E23, K23I = IN23I + I23;
  constexpr int KP4E  = ((K4E  + 63) / 64) * 64;   // 18048
  constexpr int KP4I  = ((K4I  + 63) / 64) * 64;   // 18048
  constexpr int KP23E = ((K23E + 63) / 64) * 64;   // 10816
  constexpr int KP23I = ((K23I + 63) / 64) * 64;   // 10816
  constexpr int SPLITS = 16;

  const float* x_on   = (const float*)d_in[0];
  const float* x_off  = (const float*)d_in[1];
  const float* h4e0   = (const float*)d_in[2];
  const float* h4i0   = (const float*)d_in[3];
  const float* h23e0  = (const float*)d_in[4];
  const float* h23i0  = (const float*)d_in[5];
  const float* Wih4e  = (const float*)d_in[6];  const float* Whh4e  = (const float*)d_in[7];
  const float* bih4e  = (const float*)d_in[8];  const float* bhh4e  = (const float*)d_in[9];
  const float* Wih4i  = (const float*)d_in[10]; const float* Whh4i  = (const float*)d_in[11];
  const float* bih4i  = (const float*)d_in[12]; const float* bhh4i  = (const float*)d_in[13];
  const float* Wih23e = (const float*)d_in[14]; const float* Whh23e = (const float*)d_in[15];
  const float* bih23e = (const float*)d_in[16]; const float* bhh23e = (const float*)d_in[17];
  const float* Wih23i = (const float*)d_in[18]; const float* Whh23i = (const float*)d_in[19];
  const float* bih23i = (const float*)d_in[20]; const float* bhh23i = (const float*)d_in[21];

  float* out4e  = (float*)d_out;
  float* out4i  = out4e  + (size_t)Bn * T * E4;
  float* out23e = out4i  + (size_t)Bn * T * I4;
  float* out23i = out23e + (size_t)Bn * T * E23;

  char* ws = (char*)d_ws;
  size_t off = 0;
  auto carve = [&](size_t bytes) -> char* {
    char* p = ws + off;
    off += (bytes + 255) & ~(size_t)255;
    return p;
  };
  bf16* wc4e   = (bf16*)carve((size_t)E4  * KP4E  * 2);
  bf16* wc4i   = (bf16*)carve((size_t)I4  * KP4I  * 2);
  bf16* wc23e  = (bf16*)carve((size_t)E23 * KP23E * 2);
  bf16* wc23i  = (bf16*)carve((size_t)I23 * KP23I * 2);
  float* bs4e  = (float*)carve((size_t)E4  * 4);
  float* bs4i  = (float*)carve((size_t)I4  * 4);
  float* bs23e = (float*)carve((size_t)E23 * 4);
  float* bs23i = (float*)carve((size_t)I23 * 4);
  bf16* xon_bf  = (bf16*)carve((size_t)Bn * T * X * 2);
  bf16* xoff_bf = (bf16*)carve((size_t)Bn * T * X * 2);
  bf16* h4e_bf  = (bf16*)carve((size_t)Bn * E4  * 2);
  bf16* h4i_bf  = (bf16*)carve((size_t)Bn * I4  * 2);
  bf16* h23e_bf = (bf16*)carve((size_t)Bn * E23 * 2);
  bf16* h23i_bf = (bf16*)carve((size_t)Bn * I23 * 2);
  bf16* act4e   = (bf16*)carve((size_t)Bn * KP4E  * 2);
  bf16* act4i   = (bf16*)carve((size_t)Bn * KP4I  * 2);
  bf16* act23e  = (bf16*)carve((size_t)Bn * KP23E * 2);
  bf16* act23i  = (bf16*)carve((size_t)Bn * KP23I * 2);
  float* part   = (float*)carve((size_t)SPLITS * Bn * E4 * 4);  // reused by every cell

  auto cvt = [&](bf16* d, const float* s, int n) {
    k_cvt_f32_bf16<<<cdiv(n, 256), 256, 0, stream>>>(d, s, n);
  };
  cvt(xon_bf,  x_on,  Bn * T * X);
  cvt(xoff_bf, x_off, Bn * T * X);
  cvt(h4e_bf,  h4e0,  Bn * E4);
  cvt(h4i_bf,  h4i0,  Bn * I4);
  cvt(h23e_bf, h23e0, Bn * E23);
  cvt(h23i_bf, h23i0, Bn * I23);

  auto packw = [&](bf16* d, const float* wih, const float* whh, int H, int IN, int Kpad) {
    long long tot = (long long)H * Kpad;
    k_pack_weights<<<(int)((tot + 255) / 256), 256, 0, stream>>>(d, wih, whh, H, IN, Kpad);
  };
  packw(wc4e,  Wih4e,  Whh4e,  E4,  IN4E,  KP4E);
  packw(wc4i,  Wih4i,  Whh4i,  I4,  IN4I,  KP4I);
  packw(wc23e, Wih23e, Whh23e, E23, IN23E, KP23E);
  packw(wc23i, Wih23i, Whh23i, I23, IN23I, KP23I);

  k_bias_sum<<<cdiv(E4, 256),  256, 0, stream>>>(bs4e,  bih4e,  bhh4e,  E4);
  k_bias_sum<<<cdiv(I4, 256),  256, 0, stream>>>(bs4i,  bih4i,  bhh4i,  I4);
  k_bias_sum<<<cdiv(E23, 256), 256, 0, stream>>>(bs23e, bih23e, bhh23e, E23);
  k_bias_sum<<<cdiv(I23, 256), 256, 0, stream>>>(bs23i, bih23i, bhh23i, I23);

  k_zero_bf16<<<cdiv(Bn * KP4E, 256),  256, 0, stream>>>(act4e,  Bn * KP4E);
  k_zero_bf16<<<cdiv(Bn * KP4I, 256),  256, 0, stream>>>(act4i,  Bn * KP4I);
  k_zero_bf16<<<cdiv(Bn * KP23E, 256), 256, 0, stream>>>(act23e, Bn * KP23E);
  k_zero_bf16<<<cdiv(Bn * KP23I, 256), 256, 0, stream>>>(act23i, Bn * KP23I);

  auto gemm = [&](const bf16* act, const bf16* wcat, int Kpad, int H) {
    int waves = (H / 48) * SPLITS;
    k_cell_gemm_splitk<<<cdiv(waves, 4), 128, 0, stream>>>(act, wcat, Kpad, H, SPLITS, part);
  };
  auto epi = [&](const float* bias, int H, float* outCell, int t, bf16* hstate) {
    k_cell_epilogue<<<cdiv(16 * H, 256), 256, 0, stream>>>(
        part, bias, H, SPLITS, outCell + (size_t)t * H, T * H, hstate);
  };

  for (int t = 0; t < T; ++t) {
    const bf16* xo = xon_bf  + (size_t)t * X;   // row stride T*X
    const bf16* xf = xoff_bf + (size_t)t * X;

    // L4 exc: [x_on, x_off, h4i, h23e | h4e]
    k_pack_act<<<cdiv(16 * K4E, 256), 256, 0, stream>>>(act4e, KP4E, K4E,
        xo, T * X, X,  xf, T * X, X,  h4i_bf, I4, I4,  h23e_bf, E23, E23,  h4e_bf, E4, E4);
    gemm(act4e, wc4e, KP4E, E4);
    epi(bs4e, E4, out4e, t, h4e_bf);

    // L4 inh: [x_on, x_off, h4e(new), h23e | h4i]
    k_pack_act<<<cdiv(16 * K4I, 256), 256, 0, stream>>>(act4i, KP4I, K4I,
        xo, T * X, X,  xf, T * X, X,  h4e_bf, E4, E4,  h23e_bf, E23, E23,  h4i_bf, I4, I4);
    gemm(act4i, wc4i, KP4I, I4);
    epi(bs4i, I4, out4i, t, h4i_bf);

    // L23 exc: [h4e(new), h23i | h23e]
    k_pack_act<<<cdiv(16 * K23E, 256), 256, 0, stream>>>(act23e, KP23E, K23E,
        h4e_bf, E4, E4,  h23i_bf, I23, I23,  h23e_bf, E23, E23,
        h23e_bf, 0, 0,  h23e_bf, 0, 0);
    gemm(act23e, wc23e, KP23E, E23);
    epi(bs23e, E23, out23e, t, h23e_bf);

    // L23 inh: [h4e(new), h23e(new) | h23i]
    k_pack_act<<<cdiv(16 * K23I, 256), 256, 0, stream>>>(act23i, KP23I, K23I,
        h4e_bf, E4, E4,  h23e_bf, E23, E23,  h23i_bf, I23, I23,
        h23i_bf, 0, 0,  h23i_bf, 0, 0);
    gemm(act23i, wc23i, KP23I, I23);
    epi(bs23i, I23, out23i, t, h23i_bf);
  }
}
